// CosSimAttn_43490838839488
// MI455X (gfx1250) — compile-verified, matrix operand order checked by reading, not run on hardware
//
#include <hip/hip_runtime.h>

typedef __bf16 bf16_t;
typedef __bf16 v16bf __attribute__((ext_vector_type(16)));
typedef float  v8f   __attribute__((ext_vector_type(8)));
typedef unsigned int u32x4 __attribute__((ext_vector_type(4)));

union Frag16 { u32x4 u[2]; v16bf v; };

static __device__ __forceinline__ bf16_t tobf(float f) { return (bf16_t)f; }

static __device__ __forceinline__ v8f wmma_bf16(v16bf a, v16bf b, v8f c) {
  // D = A(16x32 bf16) * B(32x16 bf16) + C(16x16 f32)
  return __builtin_amdgcn_wmma_f32_16x16x32_bf16(false, a, false, b, (short)0, c,
                                                 false, false);
}

// ---------------------------------------------------------------------------
// f32 -> bf16 elementwise convert
// ---------------------------------------------------------------------------
__global__ void cvt_f32_bf16_kernel(const float* __restrict__ src,
                                    bf16_t* __restrict__ dst, int n) {
  int i = blockIdx.x * blockDim.x + threadIdx.x;
  int stride = gridDim.x * blockDim.x;
  for (; i < n; i += stride) dst[i] = tobf(src[i]);
}

// ---------------------------------------------------------------------------
// GEMM: C(MxN) = A(MxK, bf16 row-major) * B(KxN, bf16 row-major) + bias
// workgroup tile 128x128, wave tile 32x64 (2x4 WMMA), K-step 32.
// 8 waves arranged 4 (M) x 2 (N).
// ---------------------------------------------------------------------------
#define GT_M 128
#define GT_N 128
#define GT_K 32
#define LDSA_STRIDE 48
#define LDSB_STRIDE 48

__global__ __launch_bounds__(256) void gemm_bf16_kernel(
    const bf16_t* __restrict__ A, const bf16_t* __restrict__ B,
    const float* __restrict__ bias, void* __restrict__ C,
    int M, int N, int K, int store_bf16) {
  __shared__ bf16_t ldsA[GT_M * LDSA_STRIDE];  // [m][k] padded
  __shared__ bf16_t ldsB[GT_N * LDSB_STRIDE];  // transposed: [n][k] padded

  const int t = threadIdx.x;
  const int lane = t & 31;
  const int wave = t >> 5;
  const int lane16 = lane & 15;
  const int hlf = lane >> 4;           // fragment K-half selector
  const int wm = wave & 3;             // 4 wave rows (32 each)
  const int wn = wave >> 2;            // 2 wave cols (64 each)
  const int m0 = blockIdx.y * GT_M;
  const int n0 = blockIdx.x * GT_N;

  v8f zero = {};
  v8f c[2][4];
#pragma unroll
  for (int ti = 0; ti < 2; ++ti)
#pragma unroll
    for (int tj = 0; tj < 4; ++tj) c[ti][tj] = zero;

  const int arow = t >> 1;             // 0..127
  const int aseg = (t & 1) * 16;       // 0 or 16
  const int bk = t & 31;               // k within tile
  const int bn0 = (t >> 5) * 16;       // n within tile, 16 per thread

  for (int k0 = 0; k0 < K; k0 += GT_K) {
    // --- cooperative A tile load: 128x32 row-major ---
    const bf16_t* ga = A + (size_t)(m0 + arow) * K + k0 + aseg;
    u32x4 a0 = *(const u32x4*)(ga);
    u32x4 a1 = *(const u32x4*)(ga + 8);
    *(u32x4*)(&ldsA[arow * LDSA_STRIDE + aseg]) = a0;
    *(u32x4*)(&ldsA[arow * LDSA_STRIDE + aseg + 8]) = a1;
    // --- cooperative B tile load with transpose: ldsB[n][k] ---
    const bf16_t* gb = B + (size_t)(k0 + bk) * N + n0 + bn0;
    u32x4 b0 = *(const u32x4*)(gb);
    u32x4 b1 = *(const u32x4*)(gb + 8);
    union { u32x4 u; bf16_t h[8]; } ub0, ub1;
    ub0.u = b0; ub1.u = b1;
#pragma unroll
    for (int i = 0; i < 8; ++i) {
      ldsB[(bn0 + i) * LDSB_STRIDE + bk] = ub0.h[i];
      ldsB[(bn0 + 8 + i) * LDSB_STRIDE + bk] = ub1.h[i];
    }
    // prefetch next K-slab into cache while this tile computes
    if (k0 + GT_K < K) {
      __builtin_prefetch(ga + GT_K, 0, 1);
      __builtin_prefetch(gb + (size_t)GT_K * N, 0, 1);
    }
    __syncthreads();

    // --- fragments ---
    Frag16 af[2], bfr[4];
#pragma unroll
    for (int ti = 0; ti < 2; ++ti) {
      int row = wm * 32 + ti * 16 + lane16;
      const bf16_t* p = &ldsA[row * LDSA_STRIDE + hlf * 8];
      af[ti].u[0] = *(const u32x4*)(p);       // K = hlf*8 + 0..7
      af[ti].u[1] = *(const u32x4*)(p + 16);  // K = 16 + hlf*8 + 0..7
    }
#pragma unroll
    for (int tj = 0; tj < 4; ++tj) {
      int col = wn * 64 + tj * 16 + lane16;
      const bf16_t* p = &ldsB[col * LDSB_STRIDE + hlf * 8];
      bfr[tj].u[0] = *(const u32x4*)(p);
      bfr[tj].u[1] = *(const u32x4*)(p + 16);
    }
#pragma unroll
    for (int ti = 0; ti < 2; ++ti)
#pragma unroll
      for (int tj = 0; tj < 4; ++tj)
        c[ti][tj] = wmma_bf16(af[ti].v, bfr[tj].v, c[ti][tj]);
    __syncthreads();
  }

  // --- epilogue: C layout lane holds col=lane16, rows r + hlf*8 ---
#pragma unroll
  for (int ti = 0; ti < 2; ++ti)
#pragma unroll
    for (int tj = 0; tj < 4; ++tj)
#pragma unroll
      for (int r = 0; r < 8; ++r) {
        int gm = m0 + wm * 32 + ti * 16 + r + hlf * 8;
        int gn = n0 + wn * 64 + tj * 16 + lane16;
        float v = c[ti][tj][r] + bias[gn];
        if (store_bf16)
          ((bf16_t*)C)[(size_t)gm * N + gn] = tobf(v);
        else
          ((float*)C)[(size_t)gm * N + gn] = v;
      }
}

// ---------------------------------------------------------------------------
// QKNorm + 2D image RoPE + L2 normalize; qkv (b*n, 3072) bf16 -> Q/K/V (b,h,n,64)
// ---------------------------------------------------------------------------
__global__ __launch_bounds__(256) void qkv_prep_kernel(
    const bf16_t* __restrict__ qkv, const float* __restrict__ q_gain,
    const float* __restrict__ k_gain, bf16_t* __restrict__ Q,
    bf16_t* __restrict__ K, bf16_t* __restrict__ V) {
  const int gid = blockIdx.x * blockDim.x + threadIdx.x;  // = bh*1024 + n
  const int n = gid & 1023;
  const int bh = gid >> 10;
  const int h = bh & 15;
  const int bb = bh >> 4;
  const size_t rowbase = (size_t)(bb * 1024 + n) * 3072;
  const float rowp = (float)(n >> 5);  // side = 32
  const float colp = (float)(n & 31);
  const size_t outbase = (size_t)gid * 64;

  // V: straight copy
  for (int j = 0; j < 64; ++j)
    V[outbase + j] = qkv[rowbase + 2048 + h * 64 + j];

  for (int part = 0; part < 2; ++part) {
    const float* gain = part ? k_gain : q_gain;
    bf16_t* dst = part ? K : Q;
    float x[64];
    float ss = 0.f;
#pragma unroll
    for (int j = 0; j < 64; ++j) {
      x[j] = (float)qkv[rowbase + part * 1024 + h * 64 + j];
      ss += x[j] * x[j];
    }
    float rms = rsqrtf(ss * (1.0f / 64.0f) + 1e-6f);
#pragma unroll
    for (int j = 0; j < 64; ++j) x[j] *= rms * gain[j];
    // axial RoPE: dims [0:16) row-freqs, [16:32) col-freqs, pairs (j, j+32)
#pragma unroll
    for (int j = 0; j < 32; ++j) {
      float f = (float)(j & 15) * (1.0f / 16.0f);
      float freq = __powf(10000.0f, -f);
      float ang = (j < 16 ? rowp : colp) * freq;
      float cs = __cosf(ang), sn = __sinf(ang);
      float a = x[j], bv = x[j + 32];
      x[j] = a * cs - bv * sn;
      x[j + 32] = bv * cs + a * sn;
    }
    float ss2 = 0.f;
#pragma unroll
    for (int j = 0; j < 64; ++j) ss2 += x[j] * x[j];
    float inv = rsqrtf(fmaxf(ss2, 1e-20f));
#pragma unroll
    for (int j = 0; j < 64; ++j) dst[outbase + j] = tobf(x[j] * inv);
  }
}

// ---------------------------------------------------------------------------
// Flash attention, cosine-sim scale = 8. One wave = 16 query rows;
// workgroup = 8 waves = 128 rows. Key blocks of 32 per iteration.
// ---------------------------------------------------------------------------
#define VT_STRIDE 48

__global__ __launch_bounds__(256) void attn_kernel(
    const bf16_t* __restrict__ Q, const bf16_t* __restrict__ Kk,
    const bf16_t* __restrict__ V, bf16_t* __restrict__ O) {
  __shared__ bf16_t ldsVT[64 * VT_STRIDE];   // V tile transposed: [d][key]
  __shared__ bf16_t ldsP[8 * 16 * 32];       // per-wave P staging (C->A relayout)

  const int t = threadIdx.x;
  const int lane = t & 31;
  const int wave = t >> 5;
  const int lane16 = lane & 15;
  const int hlf = lane >> 4;
  const int bh = blockIdx.y;
  const int bb = bh >> 4, h = bh & 15;
  const int qbase = blockIdx.x * 128 + wave * 16;
  const bf16_t* Qb = Q + (size_t)bh * 1024 * 64;
  const bf16_t* Kb = Kk + (size_t)bh * 1024 * 64;
  const bf16_t* Vb = V + (size_t)bh * 1024 * 64;

  // A fragments of Q: rows = lane16, two K-steps over d=64
  Frag16 aq[2];
#pragma unroll
  for (int ks = 0; ks < 2; ++ks) {
    const bf16_t* p = Qb + (size_t)(qbase + lane16) * 64 + ks * 32 + hlf * 8;
    aq[ks].u[0] = *(const u32x4*)(p);
    aq[ks].u[1] = *(const u32x4*)(p + 16);
  }

  float m_r[8], l_r[8];
  v8f zero = {};
  v8f o_acc[4];
#pragma unroll
  for (int r = 0; r < 8; ++r) { m_r[r] = -1e30f; l_r[r] = 0.f; }
#pragma unroll
  for (int tt = 0; tt < 4; ++tt) o_acc[tt] = zero;

  bf16_t* myP = &ldsP[wave * 16 * 32];
  const int vk = t & 31;
  const int vd0 = (t >> 5) * 8;

  for (int j0 = 0; j0 < 1024; j0 += 32) {
    __syncthreads();
    {  // cooperative transpose of V block (32 keys x 64 d) into ldsVT[d][key]
      u32x4 vv = *(const u32x4*)(Vb + (size_t)(j0 + vk) * 64 + vd0);
      union { u32x4 u; bf16_t hh[8]; } uu; uu.u = vv;
#pragma unroll
      for (int i = 0; i < 8; ++i) ldsVT[(vd0 + i) * VT_STRIDE + vk] = uu.hh[i];
    }
    // prefetch next key/value blocks
    if (j0 + 32 < 1024) {
      __builtin_prefetch(Vb + (size_t)(j0 + 32 + vk) * 64 + vd0, 0, 1);
      __builtin_prefetch(Kb + (size_t)(j0 + 32 + lane) * 64, 0, 1);
    }
    __syncthreads();

    // scores: Q(16x64) @ K^T -> two 16x16 tiles
    v8f s[2];
#pragma unroll
    for (int jt = 0; jt < 2; ++jt) {
      Frag16 bk0, bk1;
      const bf16_t* kp = Kb + (size_t)(j0 + jt * 16 + lane16) * 64;
      bk0.u[0] = *(const u32x4*)(kp + hlf * 8);
      bk0.u[1] = *(const u32x4*)(kp + 16 + hlf * 8);
      bk1.u[0] = *(const u32x4*)(kp + 32 + hlf * 8);
      bk1.u[1] = *(const u32x4*)(kp + 48 + hlf * 8);
      v8f sc = zero;
      sc = wmma_bf16(aq[0].v, bk0.v, sc);
      sc = wmma_bf16(aq[1].v, bk1.v, sc);
      s[jt] = sc * 8.0f;  // COS_SIM_SCALE
    }

    // online softmax: row = r + hlf*8; 16 cols live across a 16-lane half
    float mnew[8];
#pragma unroll
    for (int r = 0; r < 8; ++r) {
      float mm = fmaxf(s[0][r], s[1][r]);
#pragma unroll
      for (int d = 8; d >= 1; d >>= 1) mm = fmaxf(mm, __shfl_xor(mm, d, 32));
      mnew[r] = fmaxf(m_r[r], mm);
    }
#pragma unroll
    for (int r = 0; r < 8; ++r) {
      float alpha = __expf(m_r[r] - mnew[r]);
      l_r[r] *= alpha;
#pragma unroll
      for (int tt = 0; tt < 4; ++tt) o_acc[tt][r] *= alpha;
      m_r[r] = mnew[r];
    }
#pragma unroll
    for (int jt = 0; jt < 2; ++jt) {
#pragma unroll
      for (int r = 0; r < 8; ++r) {
        float p = __expf(s[jt][r] - mnew[r]);
        myP[(r + hlf * 8) * 32 + jt * 16 + lane16] = tobf(p);
        float ps = p;
#pragma unroll
        for (int d = 8; d >= 1; d >>= 1) ps += __shfl_xor(ps, d, 32);
        l_r[r] += ps;
      }
    }

    // P re-layout (C -> A fragment) via per-wave LDS tile
    Frag16 pa;
    {
      const bf16_t* pp = &myP[lane16 * 32 + hlf * 8];
      pa.u[0] = *(const u32x4*)(pp);
      pa.u[1] = *(const u32x4*)(pp + 16);
    }
    // o += P(16x32) @ V(32x64)
#pragma unroll
    for (int tt = 0; tt < 4; ++tt) {
      Frag16 bv;
      const bf16_t* vp = &ldsVT[(tt * 16 + lane16) * VT_STRIDE + hlf * 8];
      bv.u[0] = *(const u32x4*)(vp);
      bv.u[1] = *(const u32x4*)(vp + 16);
      o_acc[tt] = wmma_bf16(pa.v, bv.v, o_acc[tt]);
    }
  }

  // normalize and store to (b, n, dm) layout
#pragma unroll
  for (int r = 0; r < 8; ++r) {
    float inv = 1.0f / l_r[r];
    int q = qbase + r + hlf * 8;
    size_t ob = ((size_t)(bb * 1024 + q)) * 1024 + h * 64;
#pragma unroll
    for (int tt = 0; tt < 4; ++tt)
      O[ob + tt * 16 + lane16] = tobf(o_acc[tt][r] * inv);
  }
}

// ---------------------------------------------------------------------------
extern "C" void kernel_launch(void* const* d_in, const int* in_sizes, int n_in,
                              void* d_out, int out_size, void* d_ws, size_t ws_size,
                              hipStream_t stream) {
  const float* x      = (const float*)d_in[0];
  const float* w_qkv  = (const float*)d_in[1];
  const float* b_qkv  = (const float*)d_in[2];
  const float* q_gain = (const float*)d_in[3];
  const float* k_gain = (const float*)d_in[4];
  const float* w_out  = (const float*)d_in[5];
  const float* b_out  = (const float*)d_in[6];

  const size_t BN = 8 * 1024;       // b*n rows
  const size_t DM = 1024;

  char* ws = (char*)d_ws;
  bf16_t* xb    = (bf16_t*)ws; ws += BN * DM * 2;          // 16 MB
  bf16_t* wqkvb = (bf16_t*)ws; ws += DM * 3 * DM * 2;      //  6 MB
  bf16_t* woutb = (bf16_t*)ws; ws += DM * DM * 2;          //  2 MB
  bf16_t* qkvb  = (bf16_t*)ws; ws += BN * 3 * DM * 2;      // 48 MB
  bf16_t* Qd    = (bf16_t*)ws; ws += BN * DM * 2;          // 16 MB (b,h,n,64)
  bf16_t* Kd    = (bf16_t*)ws; ws += BN * DM * 2;          // 16 MB
  bf16_t* Vd    = (bf16_t*)ws; ws += BN * DM * 2;          // 16 MB
  bf16_t* Od    = (bf16_t*)ws;                             // 16 MB (b,n,dm)

  cvt_f32_bf16_kernel<<<2048, 256, 0, stream>>>(x, xb, (int)(BN * DM));
  cvt_f32_bf16_kernel<<<768, 256, 0, stream>>>(w_qkv, wqkvb, (int)(DM * 3 * DM));
  cvt_f32_bf16_kernel<<<256, 256, 0, stream>>>(w_out, woutb, (int)(DM * DM));

  // qkv = x @ w_qkv + b_qkv  (8192 x 3072 x 1024), bf16 out
  gemm_bf16_kernel<<<dim3(3072 / GT_N, 8192 / GT_M), 256, 0, stream>>>(
      xb, wqkvb, b_qkv, qkvb, 8192, 3072, 1024, 1);

  // QKNorm + RoPE + L2 norm -> Q,K,V (b,h,n,64)
  qkv_prep_kernel<<<512, 256, 0, stream>>>(qkvb, q_gain, k_gain, Qd, Kd, Vd);

  // flash cosine-sim attention -> Od (b,n,dm)
  attn_kernel<<<dim3(8, 128), 256, 0, stream>>>(Qd, Kd, Vd, Od);

  // out = Od @ w_out + b_out  (8192 x 1024 x 1024), f32 out
  gemm_bf16_kernel<<<dim3(1024 / GT_N, 8192 / GT_M), 256, 0, stream>>>(
      Od, woutb, b_out, d_out, 8192, 1024, 1024, 0);
}
